// GraphSSM_42786464203608
// MI455X (gfx1250) — matrix operand order repaired and checked by reading, not compiled, not run on hardware
//
#include <hip/hip_runtime.h>
#include <hip/hip_bf16.h>

#define D_MODEL 768
#define D_STATE 16
#define D_CONV  4
#define D_INNER 1536
#define DT_RANK 48
#define B_SZ    2
#define SEQ     2048
#define M_TOT   (B_SZ * SEQ)              // 4096
#define SSM_W   (DT_RANK + 2 * D_STATE)   // 80

typedef __bf16 v16bf __attribute__((ext_vector_type(16)));
typedef float  v8f   __attribute__((ext_vector_type(8)));

__device__ __forceinline__ unsigned int pack2_bf16(float a, float b) {
    unsigned int ua = __float_as_uint(a);
    unsigned int ub = __float_as_uint(b);
    // round-to-nearest-even f32 -> bf16
    ua = (ua + 0x7FFFu + ((ua >> 16) & 1u)) >> 16;
    ub = (ub + 0x7FFFu + ((ub >> 16) & 1u)) >> 16;
    return (ub << 16) | (ua & 0xFFFFu);
}

// ---------------------------------------------------------------------------
// Tiled WMMA GEMM:  C[M,N] = A[M,K] * W[N,K]^T  (fp32 memory, bf16 MACs,
// fp32 accumulate). Block tile 128(M) x 64(N) x 64(K); 256 threads = 8 waves
// (4 along M x 2 along N); each wave owns a 32x32 output patch and issues
// 8 v_wmma per K-stage (2 K-substeps x 2x2 tiles).
// MODE 0: plain store. MODE 1: softplus(x + bias[n]).
// CHECKED=false: tile guaranteed fully in-bounds -> branch-free float2 loads.
// ---------------------------------------------------------------------------
#define BLK_M 128
#define BLK_N 64
#define BLK_K 64
#define KP    (BLK_K / 2)     // 32 packed bf16 pairs per row

template<int MODE, bool CHECKED>
__global__ __launch_bounds__(256)
void gemm_wmma_bf16(const float* __restrict__ A, int lda,
                    const float* __restrict__ W, int ldw,
                    float* __restrict__ C, int ldc,
                    int M, int N, int K,
                    const float* __restrict__ bias)
{
    __shared__ unsigned int As[BLK_M][KP + 1];   // +1 pad vs bank conflicts
    __shared__ unsigned int Ws[BLK_N][KP + 1];

    const int tid     = threadIdx.x;
    const int lane    = tid & 31;
    const int wv      = tid >> 5;
    const int warpM   = wv & 3;        // 0..3
    const int warpN   = wv >> 2;       // 0..1
    const int blkM    = blockIdx.y * BLK_M;
    const int blkN    = blockIdx.x * BLK_N;
    const int halfsel = lane >> 4;     // 0 | 1
    const int lane15  = lane & 15;

    v8f acc[2][2];
    #pragma unroll
    for (int i = 0; i < 2; ++i)
        #pragma unroll
        for (int j = 0; j < 2; ++j)
            acc[i][j] = (v8f)0.0f;

    for (int kb = 0; kb < K; kb += BLK_K) {
        // ---- stage A tile: 128 rows x 32 pairs = 4096 pairs, 16/thread
        #pragma unroll
        for (int i = 0; i < 16; ++i) {
            int p   = tid + i * 256;
            int row = p >> 5;
            int pc  = p & 31;
            int gm  = blkM + row;
            int gk  = kb + pc * 2;
            float a0, a1;
            if (CHECKED) {
                a0 = (gm < M && gk     < K) ? A[(size_t)gm * lda + gk]     : 0.f;
                a1 = (gm < M && gk + 1 < K) ? A[(size_t)gm * lda + gk + 1] : 0.f;
            } else {
                float2 v = *(const float2*)&A[(size_t)gm * lda + gk];
                a0 = v.x; a1 = v.y;
            }
            As[row][pc] = pack2_bf16(a0, a1);
        }
        // ---- stage W tile: 64 rows x 32 pairs = 2048 pairs, 8/thread
        #pragma unroll
        for (int i = 0; i < 8; ++i) {
            int p   = tid + i * 256;
            int row = p >> 5;
            int pc  = p & 31;
            int gn  = blkN + row;
            int gk  = kb + pc * 2;
            float b0, b1;
            if (CHECKED) {
                b0 = (gn < N && gk     < K) ? W[(size_t)gn * ldw + gk]     : 0.f;
                b1 = (gn < N && gk + 1 < K) ? W[(size_t)gn * ldw + gk + 1] : 0.f;
            } else {
                float2 v = *(const float2*)&W[(size_t)gn * ldw + gk];
                b0 = v.x; b1 = v.y;
            }
            Ws[row][pc] = pack2_bf16(b0, b1);
        }
        if (!CHECKED && kb + BLK_K < K) {
            // hint next K-tile into cache (global_prefetch_b8)
            __builtin_prefetch(&A[(size_t)(blkM + (tid >> 1)) * lda + kb + BLK_K +
                                  (tid & 1) * 32], 0, 1);
            __builtin_prefetch(&W[(size_t)(blkN + (tid & 63)) * ldw + kb + BLK_K +
                                  (tid >> 6) * 16], 0, 1);
        }
        __syncthreads();

        // ---- 2 K-substeps of 32; per-lane fragment gathers per ISA layout:
        //   A(16x32): m=lane&15, pair = halfsel*4 + (v&3) + (v>=4)*8
        //   B(32x16): n=lane&15, pair = halfsel*4 + (v&3) + (v>=4)*8
        #pragma unroll
        for (int ks = 0; ks < 2; ++ks) {
            const int kofs = ks * 16;
            union Frag { unsigned int u[8]; v16bf v; };
            Frag afr[2], bfr[2];
            #pragma unroll
            for (int ms = 0; ms < 2; ++ms) {
                int m = warpM * 32 + ms * 16 + lane15;
                #pragma unroll
                for (int v = 0; v < 8; ++v) {
                    int p = kofs + halfsel * 4 + (v & 3) + ((v >> 2) * 8);
                    afr[ms].u[v] = As[m][p];
                }
            }
            #pragma unroll
            for (int ns = 0; ns < 2; ++ns) {
                int n = warpN * 32 + ns * 16 + lane15;
                #pragma unroll
                for (int v = 0; v < 8; ++v) {
                    int p = kofs + halfsel * 4 + (v & 3) + ((v >> 2) * 8);
                    bfr[ns].u[v] = Ws[n][p];
                }
            }
            #pragma unroll
            for (int ms = 0; ms < 2; ++ms)
                #pragma unroll
                for (int ns = 0; ns < 2; ++ns)
                    acc[ms][ns] = __builtin_amdgcn_wmma_f32_16x16x32_bf16(
                        false, afr[ms].v, false, bfr[ns].v,
                        (short)0, acc[ms][ns], false, false);
        }
        __syncthreads();
    }

    // ---- epilogue: C/D layout = lanes 0-15 -> M=r, lanes 16-31 -> M=r+8
    #pragma unroll
    for (int ms = 0; ms < 2; ++ms) {
        #pragma unroll
        for (int ns = 0; ns < 2; ++ns) {
            int gn = blkN + warpN * 32 + ns * 16 + lane15;
            #pragma unroll
            for (int r = 0; r < 8; ++r) {
                int gm = blkM + warpM * 32 + ms * 16 + r + halfsel * 8;
                float x = acc[ms][ns][r];
                if (MODE == 1) {
                    if (!CHECKED || gn < N) x += bias[gn];
                    x = (x > 20.f) ? x : log1pf(__expf(x));   // softplus
                }
                if (!CHECKED || (gm < M && gn < N))
                    C[(size_t)gm * ldc + gn] = x;
            }
        }
    }
}

static inline void launch_gemm(const float* A, int lda, const float* W, int ldw,
                               float* C, int ldc, int M, int N, int K,
                               const float* bias, int mode, hipStream_t stream)
{
    dim3 grid((N + BLK_N - 1) / BLK_N, (M + BLK_M - 1) / BLK_M);
    dim3 blk(256);
    bool aligned = (M % BLK_M == 0) && (N % BLK_N == 0) && (K % BLK_K == 0);
    if (aligned) {
        if (mode == 0)
            gemm_wmma_bf16<0, false><<<grid, blk, 0, stream>>>(A, lda, W, ldw, C, ldc, M, N, K, bias);
        else
            gemm_wmma_bf16<1, false><<<grid, blk, 0, stream>>>(A, lda, W, ldw, C, ldc, M, N, K, bias);
    } else {
        if (mode == 0)
            gemm_wmma_bf16<0, true><<<grid, blk, 0, stream>>>(A, lda, W, ldw, C, ldc, M, N, K, bias);
        else
            gemm_wmma_bf16<1, true><<<grid, blk, 0, stream>>>(A, lda, W, ldw, C, ldc, M, N, K, bias);
    }
}

// ---------------------------------------------------------------------------
// Causal depthwise conv1d (width 4) + bias + SiLU.
// hidden[b,d,l] = proj[b,l,d]  (first D_INNER columns of proj row).
// Output hidden_act stored [b*L, D_INNER] row-major (GEMM-friendly).
// ---------------------------------------------------------------------------
__global__ __launch_bounds__(256)
void conv_silu_kernel(const float* __restrict__ proj,
                      const float* __restrict__ conv_w,
                      const float* __restrict__ conv_b,
                      float* __restrict__ hidden_act)
{
    int d = blockIdx.x * 256 + threadIdx.x;
    int l = blockIdx.y;
    int b = blockIdx.z;
    if (d >= D_INNER) return;

    float acc = conv_b[d];
    #pragma unroll
    for (int j = 0; j < D_CONV; ++j) {
        int ls = l + j - (D_CONV - 1);
        if (ls >= 0)
            acc += conv_w[d * D_CONV + j] *
                   proj[((size_t)(b * SEQ + ls)) * (2 * D_INNER) + d];
    }
    float s = acc / (1.f + __expf(-acc));   // SiLU
    hidden_act[((size_t)(b * SEQ + l)) * D_INNER + d] = s;
}

// ---------------------------------------------------------------------------
// Selective-scan (suffix recurrence g[s] = fin[s] + w[s+1]*g[s+1]) + output
// contraction over n, skip term, SiLU gating. One block handles (b, 16 d's);
// threads: tid = dl*16 + n -> each 16-lane half-wave holds one d with n on
// the 4 low lane bits, so __shfl_xor(1|2|4|8) reduces over n only (wave32).
// ---------------------------------------------------------------------------
__global__ __launch_bounds__(256)
void scan_kernel(const float* __restrict__ dt_full,
                 const float* __restrict__ hidden_act,
                 const float* __restrict__ ssm,
                 const float* __restrict__ proj,
                 const float* __restrict__ A_log,
                 const float* __restrict__ Dv,
                 float* __restrict__ y)
{
    const int tid = threadIdx.x;
    const int n   = tid & 15;
    const int dl  = tid >> 4;
    const int d   = blockIdx.x * 16 + dl;
    const int b   = blockIdx.y;

    const float Ac = -__expf(A_log[d * D_STATE + n]);
    const float Dd = Dv[d];

    float g = 0.f, wprev = 0.f;
    for (int s = SEQ - 1; s >= 0; --s) {
        size_t row = (size_t)(b * SEQ + s);
        float dtv = dt_full[row * D_INNER + d];
        float hv  = hidden_act[row * D_INNER + d];
        float Bv  = ssm[row * SSM_W + DT_RANK + n];
        float Cv  = ssm[row * SSM_W + DT_RANK + D_STATE + n];

        g     = dtv * Bv * hv + wprev * g;   // uses w[s+1] from prev iter
        wprev = __expf(Ac * dtv);            // becomes w[s] for next iter

        float c = 1.3f * g * Cv;
        c += __shfl_xor(c, 1, 32);
        c += __shfl_xor(c, 2, 32);
        c += __shfl_xor(c, 4, 32);
        c += __shfl_xor(c, 8, 32);

        if (n == 0) {
            float gate = proj[row * (2 * D_INNER) + D_INNER + d];
            float sg   = gate / (1.f + __expf(-gate));
            y[row * D_INNER + d] = (c + hv * Dd) * sg;
        }
    }
}

// ---------------------------------------------------------------------------
extern "C" void kernel_launch(void* const* d_in, const int* in_sizes, int n_in,
                              void* d_out, int out_size, void* d_ws, size_t ws_size,
                              hipStream_t stream)
{
    const float* input_states = (const float*)d_in[0];
    const float* in_proj_w    = (const float*)d_in[1];
    const float* conv_w       = (const float*)d_in[2];
    const float* conv_b       = (const float*)d_in[3];
    const float* x_proj_w     = (const float*)d_in[4];
    const float* dt_proj_w    = (const float*)d_in[5];
    const float* dt_proj_b    = (const float*)d_in[6];
    const float* A_log        = (const float*)d_in[7];
    const float* Dvec         = (const float*)d_in[8];
    const float* out_proj_w   = (const float*)d_in[9];
    // d_in[10] = context_len (always 2, chain-scan branch) -- unused

    float* ws     = (float*)d_ws;
    float* proj   = ws;                                        // [4096, 3072]
    float* hidden = proj   + (size_t)M_TOT * 2 * D_INNER;      // [4096, 1536]
    float* ssm    = hidden + (size_t)M_TOT * D_INNER;          // [4096,   80]
    float* dtf    = ssm    + (size_t)M_TOT * SSM_W;            // [4096, 1536]
    float* ybuf   = dtf    + (size_t)M_TOT * D_INNER;          // [4096, 1536]

    dim3 blk(256);

    // 1. in_proj: proj = X * in_proj_w^T   [4096 x 3072]  (fast path)
    launch_gemm(input_states, D_MODEL, in_proj_w, D_MODEL,
                proj, 2 * D_INNER, M_TOT, 2 * D_INNER, D_MODEL,
                nullptr, 0, stream);

    // 2. depthwise causal conv + SiLU
    conv_silu_kernel<<<dim3(D_INNER / 256, SEQ, B_SZ), blk, 0, stream>>>(
        proj, conv_w, conv_b, hidden);

    // 3. x_proj: ssm = hidden * x_proj_w^T   [4096 x 80]  (checked: N=80)
    launch_gemm(hidden, D_INNER, x_proj_w, D_INNER,
                ssm, SSM_W, M_TOT, SSM_W, D_INNER,
                nullptr, 0, stream);

    // 4. dt_proj + softplus: dtf = softplus(ssm[:, :48]*dt_proj_w^T + b)
    //    (checked: K=48)
    launch_gemm(ssm, SSM_W, dt_proj_w, DT_RANK,
                dtf, D_INNER, M_TOT, D_INNER, DT_RANK,
                dt_proj_b, 1, stream);

    // 5. selective scan + C-contraction + skip + gating -> ybuf
    scan_kernel<<<dim3(D_INNER / 16, B_SZ), blk, 0, stream>>>(
        dtf, hidden, ssm, proj, A_log, Dvec, ybuf);

    // 6. out_proj: out = ybuf * out_proj_w^T   [4096 x 768]  (fast path)
    launch_gemm(ybuf, D_INNER, out_proj_w, D_INNER,
                (float*)d_out, D_MODEL, M_TOT, D_MODEL, D_INNER,
                nullptr, 0, stream);
}
